// SparseMoE_32315333935434
// MI455X (gfx1250) — matrix operand order repaired and checked by reading, compiled, not run on hardware
//
#include <hip/hip_runtime.h>
#include <hip/hip_bf16.h>
#include <cmath>

// ---------------- problem constants ----------------
constexpr int Hd = 1024;   // hidden
constexpr int Fd = 4096;   // ffn
constexpr int Ed = 8;      // experts
constexpr int Td = 4096;   // tokens (2*2048)
constexpr int TM = 16;     // tokens per MoE block
constexpr int TILES = Td / TM;

// ---------------- WMMA types ----------------
typedef __attribute__((ext_vector_type(16))) __bf16 v16bf;
typedef __attribute__((ext_vector_type(8)))  float  v8f;
typedef __attribute__((vector_size(16)))     int    v4i_128;   // b128 payload type

union FragU { uint4 q[2]; v16bf v; };

__device__ __forceinline__ uint32_t bf16rne(float f) {
    uint32_t u = __float_as_uint(f);
    return (u + 0x7FFFu + ((u >> 16) & 1u)) >> 16;
}
__device__ __forceinline__ uint32_t pack2(float a, float b) {
    return bf16rne(a) | (bf16rne(b) << 16);
}

// ---------------- CDNA5 async global->LDS copy (ASYNCcnt path) --------------
#if defined(__has_builtin)
#if __has_builtin(__builtin_amdgcn_global_load_async_to_lds_b128)
#define HAVE_ASYNC_LDS 1
#endif
#endif

__device__ __forceinline__ void async_cp16(const void* g, void* l) {
#if defined(HAVE_ASYNC_LDS)
    __builtin_amdgcn_global_load_async_to_lds_b128(
        (__attribute__((address_space(1))) v4i_128*)g,
        (__attribute__((address_space(3))) v4i_128*)l, 0, 0);
#else
    *(uint4*)l = *(const uint4*)g;   // sync fallback
#endif
}
__device__ __forceinline__ void wait_async0() {
#if defined(__has_builtin)
#if __has_builtin(__builtin_amdgcn_s_wait_asynccnt)
    __builtin_amdgcn_s_wait_asynccnt(0);
#else
    asm volatile("s_wait_asynccnt 0x0" ::: "memory");
#endif
#else
    asm volatile("s_wait_asynccnt 0x0" ::: "memory");
#endif
}

// A fragment (16xK, 16-bit): lane<16 -> K {k0..k0+7, k0+16..k0+23},
// lane>=16 -> K {k0+8..k0+15, k0+24..k0+31}; row = lane&15.
__device__ __forceinline__ v16bf load_fragA(const uint32_t* mat, int rowstride_u32,
                                            int k0, int rsel, int ksel) {
    FragU f;
    const uint32_t* p = mat + rsel * rowstride_u32 + (k0 >> 1) + ksel * 4;
    f.q[0] = *(const uint4*)(p);
    f.q[1] = *(const uint4*)(p + 8);
    return f.v;
}
// B fragment (Kx16 staged transposed: N rows x 32 K bf16 = 16 u32/row):
// lanes 0-15 hold K {0..7,16..23}, lanes 16-31 hold K {8..15,24..31}; col = lane&15.
__device__ __forceinline__ v16bf load_fragB(const uint32_t* Ws, int nloc0,
                                            int rsel, int ksel) {
    FragU f;
    const uint32_t* p = Ws + (nloc0 + rsel) * 16 + ksel * 4;
    f.q[0] = *(const uint4*)(p);
    f.q[1] = *(const uint4*)(p + 8);
    return f.v;
}

// ---------------- 1) fp32 -> packed bf16 convert (x) ----------------
__global__ __launch_bounds__(256) void cvt_kernel(const float* __restrict__ x,
                                                  uint32_t* __restrict__ xb) {
    int i = blockIdx.x * 256 + threadIdx.x;
    float4 f = ((const float4*)x)[i];
    xb[2 * i]     = pack2(f.x, f.y);
    xb[2 * i + 1] = pack2(f.z, f.w);
}

// ------------ 1b) weight transpose + convert: W[K,N] f32 -> WT[N,K] bf16 -----
__global__ __launch_bounds__(256) void wtrans_kernel(const float* __restrict__ src,
                                                     uint32_t* __restrict__ dst,
                                                     int K, int N) {
    __shared__ unsigned short t[64][66];          // padded vs bank conflicts
    const float* s = src + (size_t)blockIdx.z * K * N;
    uint32_t* d32 = dst + (size_t)blockIdx.z * ((size_t)N * K / 2);
    int n0 = blockIdx.x * 64, k0 = blockIdx.y * 64;
    int tid = threadIdx.x;
    int nn = tid & 63, kq = tid >> 6;             // kq 0..3
    #pragma unroll
    for (int i = 0; i < 16; ++i) {
        int kk = (i << 2) + kq;
        t[nn][kk] = (unsigned short)bf16rne(s[(size_t)(k0 + kk) * N + (n0 + nn)]);
    }
    __syncthreads();
    #pragma unroll
    for (int i = 0; i < 8; ++i) {
        int lin = (i << 8) + tid;                 // 0..2047
        int rn = lin >> 5, kw = lin & 31;         // 64 rows x 32 u32
        uint32_t lo = t[rn][2 * kw], hi = t[rn][2 * kw + 1];
        d32[(size_t)(n0 + rn) * (K >> 1) + (k0 >> 1) + kw] = lo | (hi << 16);
    }
}

// ---------------- 2) gating ----------------
__global__ __launch_bounds__(256) void gate_kernel(const float* __restrict__ x,
                                                   const float* __restrict__ wg,
                                                   float* __restrict__ probs,
                                                   int* __restrict__ cnt1,
                                                   int* __restrict__ ecnt,
                                                   int* __restrict__ tlist,
                                                   float* __restrict__ gwl) {
    __shared__ float wgs[Hd * Ed];
    int tid = threadIdx.x;
    for (int i = tid; i < Hd * Ed; i += 256) wgs[i] = wg[i];
    __syncthreads();

    int wid = tid >> 5, lane = tid & 31;
    int t = blockIdx.x * 8 + wid;
    const float* xr = x + (size_t)t * Hd;

    float acc[Ed] = {};
    for (int k = lane; k < Hd; k += 32) {
        float xv = xr[k];
        #pragma unroll
        for (int e = 0; e < Ed; ++e) acc[e] += xv * wgs[k * Ed + e];
    }
    #pragma unroll
    for (int e = 0; e < Ed; ++e)
        #pragma unroll
        for (int off = 16; off > 0; off >>= 1)
            acc[e] += __shfl_down(acc[e], off, 32);

    if (lane == 0) {
        float m = acc[0];
        #pragma unroll
        for (int e = 1; e < Ed; ++e) m = fmaxf(m, acc[e]);
        float p[Ed], s = 0.f;
        #pragma unroll
        for (int e = 0; e < Ed; ++e) { p[e] = expf(acc[e] - m); s += p[e]; }
        float inv = 1.f / s;
        #pragma unroll
        for (int e = 0; e < Ed; ++e) { p[e] *= inv; probs[t * Ed + e] = p[e]; }
        int i1 = 0;
        #pragma unroll
        for (int e = 1; e < Ed; ++e) if (p[e] > p[i1]) i1 = e;
        int i2 = (i1 == 0) ? 1 : 0;
        #pragma unroll
        for (int e = 0; e < Ed; ++e) if (e != i1 && p[e] > p[i2]) i2 = e;
        float d = p[i1] + p[i2] + 1e-9f;
        atomicAdd(&cnt1[i1], 1);
        int pos1 = atomicAdd(&ecnt[i1], 1);
        tlist[i1 * Td + pos1] = t; gwl[i1 * Td + pos1] = p[i1] / d;
        int pos2 = atomicAdd(&ecnt[i2], 1);
        tlist[i2 * Td + pos2] = t; gwl[i2 * Td + pos2] = p[i2] / d;
    }
}

// ---------------- 3) aux loss (deterministic) ----------------
__global__ __launch_bounds__(256) void aux_kernel(const float* __restrict__ probs,
                                                  const int* __restrict__ cnt1,
                                                  float* __restrict__ out_aux) {
    __shared__ float red[256];
    __shared__ float Pe[Ed];
    int tid = threadIdx.x;
    for (int e = 0; e < Ed; ++e) {
        float s = 0.f;
        for (int t = tid; t < Td; t += 256) s += probs[t * Ed + e];
        red[tid] = s;
        __syncthreads();
        for (int off = 128; off > 0; off >>= 1) {
            if (tid < off) red[tid] += red[tid + off];
            __syncthreads();
        }
        if (tid == 0) Pe[e] = red[0] * (1.f / (float)Td);
        __syncthreads();
    }
    if (tid == 0) {
        float aux = 0.f;
        for (int e = 0; e < Ed; ++e)
            aux += ((float)cnt1[e] * (1.f / (float)Td)) * Pe[e];
        out_aux[0] = (float)Ed * aux;
    }
}

// ---------------- 4) routed expert FFN: async-staged bf16 WMMA ---------------
// LDS (u32): Xs[16][512] | Ws0[256][16] | Ws1[256][16] | Hs(16x4096 bf16) | tok/gw/vld
constexpr int XS_OFF = 0;
constexpr int WS0_OFF = 8192;
constexpr int WS1_OFF = 12288;
constexpr int HS_OFF = 16384;
constexpr int TK_OFF = 49152;
constexpr int SMEM_U32 = 49200;
constexpr int SMEM_BYTES = SMEM_U32 * 4;          // 196,800 B (< 320 KB WGP LDS)

// stage one slab: 256 rows (n) x 32 K bf16, 4 x b128 async per thread
__device__ __forceinline__ void issue_slab(const uint32_t* wT_e_u32, int Kdim,
                                           int nc, int k0, uint32_t* buf, int tid) {
    const uint32_t* g = wT_e_u32 + (size_t)(nc + tid) * (Kdim >> 1) + (k0 >> 1);
    uint32_t* l = buf + tid * 16;
    #pragma unroll
    for (int i = 0; i < 4; ++i)
        async_cp16((const void*)(g + i * 4), (void*)(l + i * 4));
}

__global__ __launch_bounds__(256) void moe_kernel(const uint32_t* __restrict__ xb,
                                                  const uint32_t* __restrict__ wfcT,
                                                  const float* __restrict__ bfc,
                                                  const uint32_t* __restrict__ wprT,
                                                  const float* __restrict__ bpr,
                                                  const int* __restrict__ ecnt,
                                                  const int* __restrict__ tlist,
                                                  const float* __restrict__ gwl,
                                                  float* __restrict__ out) {
    extern __shared__ uint32_t sm[];
    uint32_t* Xs  = sm + XS_OFF;
    uint32_t* Ws0 = sm + WS0_OFF;
    uint32_t* Ws1 = sm + WS1_OFF;
    uint32_t* Hs  = sm + HS_OFF;
    int*   tokL = (int*)(sm + TK_OFF);
    float* gwL  = (float*)(sm + TK_OFF + 16);
    int*   vldL = (int*)(sm + TK_OFF + 32);

    const int e    = blockIdx.x >> 8;
    const int tile = blockIdx.x & 255;
    const int cnt  = ecnt[e];
    if (tile * TM >= cnt) return;

    const int tid = threadIdx.x;
    if (tid < TM) {
        int idx = tile * TM + tid;
        int v = (idx < cnt) ? 1 : 0;
        tokL[tid] = v ? tlist[e * Td + idx] : 0;   // invalid rows alias token 0
        gwL[tid]  = v ? gwl[e * Td + idx] : 0.f;
        vldL[tid] = v;
    }
    __syncthreads();

    const uint32_t* wfcT_e = wfcT + (size_t)e * (Fd * (Hd / 2));
    const uint32_t* wprT_e = wprT + (size_t)e * (Hd * (Fd / 2));

    // async gather of 16 token rows (16 x 512 u32) into Xs
    #pragma unroll
    for (int i = 0; i < 8; ++i) {
        int lin = (i << 8) + tid;                 // 0..2047 b128 transfers
        int r = lin >> 7, c = lin & 127;          // row, 16B chunk
        async_cp16((const void*)(xb + (size_t)tokL[r] * 512 + c * 4),
                   (void*)(Xs + r * 512 + c * 4));
    }
    issue_slab(wfcT_e, Hd, 0, 0, Ws0, tid);       // prologue: GEMM1 slab (0,0)

    const int lane = tid & 31, wid = tid >> 5;
    const int rsel = lane & 15, ksel = lane >> 4;
    unsigned short* hs16 = (unsigned short*)Hs;

    // ---- GEMM1: h[16,4096] = gelu(X @ Wfc + b) ----
    for (int nc = 0; nc < Fd; nc += 256) {
        v8f c0 = {}, c1 = {};
        for (int ks = 0; ks < 32; ++ks) {         // 32 slabs (even count)
            wait_async0();
            __syncthreads();
            uint32_t* buf = (ks & 1) ? Ws1 : Ws0;
            v16bf a  = load_fragA(Xs, 512, ks * 32, rsel, ksel);
            v16bf b0 = load_fragB(buf, wid * 32, rsel, ksel);
            v16bf b1 = load_fragB(buf, wid * 32 + 16, rsel, ksel);
            uint32_t* nbuf = (ks & 1) ? Ws0 : Ws1;
            if (ks < 31)              issue_slab(wfcT_e, Hd, nc, (ks + 1) * 32, nbuf, tid);
            else if (nc + 256 < Fd)   issue_slab(wfcT_e, Hd, nc + 256, 0, Ws0, tid);
            else                      issue_slab(wprT_e, Fd, 0, 0, Ws0, tid); // GEMM2 prologue
            c0 = __builtin_amdgcn_wmma_f32_16x16x32_bf16(false, a, false, b0,
                                                         (short)0, c0, false, false);
            c1 = __builtin_amdgcn_wmma_f32_16x16x32_bf16(false, a, false, b1,
                                                         (short)0, c1, false, false);
        }
        // bias + exact GELU -> bf16 h in LDS (overlaps next-slab async copies)
        #pragma unroll
        for (int sub = 0; sub < 2; ++sub) {
            v8f cc = sub ? c1 : c0;
            int nbl = nc + wid * 32 + sub * 16 + rsel;
            float bv = bfc[e * Fd + nbl];
            #pragma unroll
            for (int v = 0; v < 8; ++v) {
                int r = v + ksel * 8;
                float h = cc[v] + bv;
                float g = 0.5f * h * (1.0f + erff(h * 0.70710678118f));
                hs16[r * Fd + nbl] = (unsigned short)bf16rne(g);
            }
        }
    }
    __syncthreads();                              // Hs visible to all waves

    // ---- GEMM2: y[16,1024] = h @ Wproj + b; out += gate*y ----
    for (int nc = 0; nc < Hd; nc += 256) {
        v8f c0 = {}, c1 = {};
        for (int ks = 0; ks < 128; ++ks) {        // 128 slabs (even count)
            wait_async0();
            __syncthreads();
            uint32_t* buf = (ks & 1) ? Ws1 : Ws0;
            v16bf a  = load_fragA(Hs, 2048, ks * 32, rsel, ksel);
            v16bf b0 = load_fragB(buf, wid * 32, rsel, ksel);
            v16bf b1 = load_fragB(buf, wid * 32 + 16, rsel, ksel);
            uint32_t* nbuf = (ks & 1) ? Ws0 : Ws1;
            if (ks < 127)             issue_slab(wprT_e, Fd, nc, (ks + 1) * 32, nbuf, tid);
            else if (nc + 256 < Hd)   issue_slab(wprT_e, Fd, nc + 256, 0, Ws0, tid);
            c0 = __builtin_amdgcn_wmma_f32_16x16x32_bf16(false, a, false, b0,
                                                         (short)0, c0, false, false);
            c1 = __builtin_amdgcn_wmma_f32_16x16x32_bf16(false, a, false, b1,
                                                         (short)0, c1, false, false);
        }
        #pragma unroll
        for (int sub = 0; sub < 2; ++sub) {
            v8f cc = sub ? c1 : c0;
            int nbl = nc + wid * 32 + sub * 16 + rsel;
            float bv = bpr[e * Hd + nbl];
            #pragma unroll
            for (int v = 0; v < 8; ++v) {
                int r = v + ksel * 8;
                if (vldL[r]) {
                    float y = (cc[v] + bv) * gwL[r];
                    atomicAdd(out + (size_t)tokL[r] * Hd + nbl, y); // 2 adds/elem -> deterministic
                }
            }
        }
    }
}

// ---------------- launcher ----------------
extern "C" void kernel_launch(void* const* d_in, const int* in_sizes, int n_in,
                              void* d_out, int out_size, void* d_ws, size_t ws_size,
                              hipStream_t stream) {
    const float* x   = (const float*)d_in[0];
    const float* wg  = (const float*)d_in[1];
    const float* wfc = (const float*)d_in[2];
    const float* bfc = (const float*)d_in[3];
    const float* wpr = (const float*)d_in[4];
    const float* bpr = (const float*)d_in[5];
    float* out = (float*)d_out;

    char* ws = (char*)d_ws;
    uint32_t* xb   = (uint32_t*)(ws);                        // 8 MB bf16 x
    uint32_t* wfcT = (uint32_t*)(ws + 8388608);              // 64 MB bf16 Wfc^T [E][F][H]
    uint32_t* wprT = (uint32_t*)(ws + 75497472);             // 64 MB bf16 Wproj^T [E][H][F]
    float* probs   = (float*)(ws + 142606336);               // Td*Ed
    int* cnt1      = (int*)(ws + 142737408);
    int* ecnt      = (int*)(ws + 142737440);
    int* tlist     = (int*)(ws + 142737472);                 // Ed*Td
    float* gwl     = (float*)(ws + 142868544);               // Ed*Td

    (void)hipMemsetAsync(d_out, 0, (size_t)out_size * sizeof(float), stream);
    (void)hipMemsetAsync(cnt1, 0, 64, stream);

    cvt_kernel<<<(Td * Hd / 4) / 256, 256, 0, stream>>>(x, xb);
    wtrans_kernel<<<dim3(Fd / 64, Hd / 64, Ed), 256, 0, stream>>>(wfc, wfcT, Hd, Fd);
    wtrans_kernel<<<dim3(Hd / 64, Fd / 64, Ed), 256, 0, stream>>>(wpr, wprT, Fd, Hd);
    gate_kernel<<<Td / 8, 256, 0, stream>>>(x, wg, probs, cnt1, ecnt, tlist, gwl);
    aux_kernel<<<1, 256, 0, stream>>>(probs, cnt1, out + (size_t)Td * Hd);

    (void)hipFuncSetAttribute((const void*)moe_kernel,
                              hipFuncAttributeMaxDynamicSharedMemorySize, SMEM_BYTES);
    moe_kernel<<<Ed * TILES, 256, SMEM_BYTES, stream>>>(xb, wfcT, bfc, wprT, bpr,
                                                        ecnt, tlist, gwl, out);
}
